// RGCNLayer_25572235280738
// MI455X (gfx1250) — compile-verified, matrix operand order checked by reading, not run on hardware
//
#include <hip/hip_runtime.h>
#include <hip/hip_bf16.h>

// ---------------------------------------------------------------------------
// RGCN layer, fused edge-tile schedule for MI455X (gfx1250, wave32).
//   msg_e = h[src_e] @ W[rel_e]   -> 16-edge x 128 tiles via v_wmma_f32_16x16x32_bf16
//   h'_v  = relu(mean msg + bias) -> f32 atomics + finalize kernel
// f32 inputs are split into bf16 hi/lo planes; C = Ah*Bh + Ah*Bl + Al*Bh.
// Software-pipelined: next group's row gather overlaps current group's WMMAs.
// ---------------------------------------------------------------------------

typedef __attribute__((ext_vector_type(16))) __bf16 v16bf;
typedef __attribute__((ext_vector_type(8)))  float  v8f;

#define DIN   128
#define DOUT  128
#define NREL  16
#define CHUNK 512     // edges binned per workgroup
#define THREADS 256   // 8 waves; wave w owns output columns [w*16, w*16+16)

// ---- split f32 -> (bf16 hi, bf16 lo) planes for h ---------------------------
__global__ void k_split_h(const float* __restrict__ h,
                          __bf16* __restrict__ hhi, __bf16* __restrict__ hlo,
                          int total) {
  int i = blockIdx.x * blockDim.x + threadIdx.x;
  if (i >= total) return;
  float x = h[i];
  __bf16 a = (__bf16)x;
  hhi[i] = a;
  hlo[i] = (__bf16)(x - (float)a);
}

// ---- split + swizzle W into per-lane WMMA B-fragment layout ----------------
// slot(r, ntile, kt, lane) holds the 16 bf16 B-matrix elements lane needs for
// v_wmma_f32_16x16x32_bf16 at K-tile kt, N-tile ntile of relation r:
//   N = ntile*16 + (lane&15)
//   koff = kt*32 + (lane>=16 ? 8 : 0)
//   elems 0..7  -> K = koff + i          (frag VGPRs 0-3)
//   elems 8..15 -> K = koff + 16 + (i-8) (frag VGPRs 4-7)
__global__ void k_split_weight(const float* __restrict__ W,
                               __bf16* __restrict__ whi, __bf16* __restrict__ wlo) {
  int slot = blockIdx.x * blockDim.x + threadIdx.x;   // NREL*8*4*32 slots
  if (slot >= NREL * 8 * 4 * 32) return;
  int lane = slot & 31;
  int kt   = (slot >> 5) & 3;
  int nt   = (slot >> 7) & 7;
  int r    = slot >> 10;
  int n    = nt * 16 + (lane & 15);
  int koff = kt * 32 + ((lane & 16) ? 8 : 0);
  __bf16* ph = whi + (size_t)slot * 16;
  __bf16* pl = wlo + (size_t)slot * 16;
#pragma unroll
  for (int i = 0; i < 16; ++i) {
    int k = koff + (i < 8 ? i : 8 + i);   // i>=8 -> koff+16+(i-8)
    float x = W[((size_t)r * DIN + k) * DOUT + n];
    __bf16 hi = (__bf16)x;
    ph[i] = hi;
    pl[i] = (__bf16)(x - (float)hi);
  }
}

// ---- zero accumulators ------------------------------------------------------
__global__ void k_zero(float* __restrict__ out, int n_out,
                       float* __restrict__ deg, int n_deg) {
  int i = blockIdx.x * blockDim.x + threadIdx.x;
  if (i < n_out) out[i] = 0.0f;
  if (i < n_deg) deg[i] = 0.0f;
}

// ---- fused message kernel ---------------------------------------------------
__global__ __launch_bounds__(THREADS)
void k_rgcn(const __bf16* __restrict__ hhi, const __bf16* __restrict__ hlo,
            const __bf16* __restrict__ whi, const __bf16* __restrict__ wlo,
            const int* __restrict__ src, const int* __restrict__ dst,
            const int* __restrict__ rel,
            float* __restrict__ out, float* __restrict__ deg, int E) {
  __shared__ int bin_cnt[NREL];
  __shared__ int bin_edge[NREL][CHUNK];
  __shared__ alignas(16) __bf16 a_hi[16][DIN];
  __shared__ alignas(16) __bf16 a_lo[16][DIN];
  __shared__ alignas(16) int g_dst[16];

  const int tid  = threadIdx.x;
  const int lane = tid & 31;
  const int wave = tid >> 5;        // 0..7 = output N-tile
  const int half = lane >> 4;       // 0 | 1
  const int lm   = lane & 15;
  const int n0   = wave * 16;
  const int base = blockIdx.x * CHUNK;
  const int row  = tid >> 4;        // staging: row 0..15
  const int cs   = (tid & 15) * 8;  // staging: 8 bf16 (16B) per thread per plane

  if (tid < NREL) bin_cnt[tid] = 0;
  __syncthreads();

  // bin edges by relation; count in-degree
  for (int i = tid; i < CHUNK; i += THREADS) {
    int e = base + i;
    if (e < E) {
      int r = rel[e];
      int p = atomicAdd(&bin_cnt[r], 1);
      bin_edge[r][p] = e;
      atomicAdd(&deg[dst[e]], 1.0f);
    }
  }
  __syncthreads();

  for (int r = 0; r < NREL; ++r) {
    int c = bin_cnt[r];               // uniform across workgroup
    if (c == 0) continue;

    // B fragments for this relation live in VGPRs across all its groups
    v16bf bh[4], bl[4];
#pragma unroll
    for (int kt = 0; kt < 4; ++kt) {
      size_t slot = (((size_t)r * 8 + wave) * 4 + kt) * 32 + lane;
      bh[kt] = *(const v16bf*)(whi + slot * 16);
      bl[kt] = *(const v16bf*)(wlo + slot * 16);
    }

    // prefetch group 0 rows into registers
    uint4 ph, pl;
    int pd;
    {
      int gi = row;
      int e  = bin_edge[r][gi < c ? gi : 0];
      int s  = src[e];
      ph = *(const uint4*)(hhi + (size_t)s * DIN + cs);
      pl = *(const uint4*)(hlo + (size_t)s * DIN + cs);
      pd = dst[e];
    }

    for (int g = 0; g < c; g += 16) {
      __syncthreads();                // previous group's A fully consumed
      // stage prefetched rows into LDS
      *(uint4*)&a_hi[row][cs] = ph;
      *(uint4*)&a_lo[row][cs] = pl;
      if ((tid & 15) == 0) g_dst[row] = pd;
      __syncthreads();

      // software pipeline: gather next group's rows while WMMAs run
      int gn = g + 16;
      if (gn < c) {
        int gi = gn + row;
        int e  = bin_edge[r][gi < c ? gi : gn];
        int s  = src[e];
        ph = *(const uint4*)(hhi + (size_t)s * DIN + cs);
        pl = *(const uint4*)(hlo + (size_t)s * DIN + cs);
        pd = dst[e];
      }

      // scatter targets: 8 dst ids as two b128 LDS loads (can't be re-split)
      uint4 d0 = *(const uint4*)&g_dst[half * 8];
      uint4 d1 = *(const uint4*)&g_dst[half * 8 + 4];
      int sd[8] = {(int)d0.x, (int)d0.y, (int)d0.z, (int)d0.w,
                   (int)d1.x, (int)d1.y, (int)d1.z, (int)d1.w};

      // preload all A fragments, then run the WMMA chain
      union { v16bf v; uint4 q[2]; } uah[4], ual[4];
#pragma unroll
      for (int kt = 0; kt < 4; ++kt) {
        int ko = kt * 32 + half * 8;
        uah[kt].q[0] = *(const uint4*)&a_hi[lm][ko];
        uah[kt].q[1] = *(const uint4*)&a_hi[lm][ko + 16];
        ual[kt].q[0] = *(const uint4*)&a_lo[lm][ko];
        ual[kt].q[1] = *(const uint4*)&a_lo[lm][ko + 16];
      }

      // C[16x16] += A[16x128] * B[128x16], bf16 split (hi*hi + hi*lo + lo*hi)
      v8f acc = {};
#pragma unroll
      for (int kt = 0; kt < 4; ++kt) {
        acc = __builtin_amdgcn_wmma_f32_16x16x32_bf16(false, uah[kt].v, false, bh[kt],
                                                      (short)0, acc, false, false);
        acc = __builtin_amdgcn_wmma_f32_16x16x32_bf16(false, uah[kt].v, false, bl[kt],
                                                      (short)0, acc, false, false);
        acc = __builtin_amdgcn_wmma_f32_16x16x32_bf16(false, ual[kt].v, false, bh[kt],
                                                      (short)0, acc, false, false);
      }

      // scatter: C/D layout -> VGPR v, lanes<16: M=v, lanes>=16: M=v+8, N=lane&15
      if (gn <= c) {
        // full group (uniform fast path): no per-element predication
#pragma unroll
        for (int v = 0; v < 8; ++v) {
          atomicAdd(&out[(size_t)sd[v] * DOUT + n0 + lm], acc[v]);
        }
      } else {
        // tail group: row M valid iff g + M < c
#pragma unroll
        for (int v = 0; v < 8; ++v) {
          int M = half * 8 + v;
          if (g + M < c) {
            atomicAdd(&out[(size_t)sd[v] * DOUT + n0 + lm], acc[v]);
          }
        }
      }
    }
  }
}

// ---- finalize: mean + bias + relu ------------------------------------------
__global__ void k_final(float* __restrict__ out, const float* __restrict__ deg,
                        const float* __restrict__ bias, int total) {
  int i = blockIdx.x * blockDim.x + threadIdx.x;
  if (i >= total) return;
  int v = i >> 7;                     // / DOUT
  float d = deg[v];
  d = d > 1.0f ? d : 1.0f;
  float x = out[i] / d + bias[i & (DOUT - 1)];
  out[i] = x > 0.0f ? x : 0.0f;
}

extern "C" void kernel_launch(void* const* d_in, const int* in_sizes, int n_in,
                              void* d_out, int out_size, void* d_ws, size_t ws_size,
                              hipStream_t stream) {
  const float* h    = (const float*)d_in[0];
  const float* W    = (const float*)d_in[1];
  const float* bias = (const float*)d_in[2];
  const int*   src  = (const int*)d_in[3];
  const int*   dst  = (const int*)d_in[4];
  const int*   rel  = (const int*)d_in[5];
  float* out = (float*)d_out;

  const int N = in_sizes[0] / DIN;    // 50000
  const int E = in_sizes[3];          // 600000

  // workspace layout (≈27 MB): deg | W_hi | W_lo | h_hi | h_lo
  char* ws = (char*)d_ws;
  float* deg = (float*)ws;
  size_t off = ((size_t)N * 4 + 255) & ~(size_t)255;
  __bf16* whi = (__bf16*)(ws + off); off += (size_t)NREL * DIN * DOUT * 2;
  __bf16* wlo = (__bf16*)(ws + off); off += (size_t)NREL * DIN * DOUT * 2;
  __bf16* hhi = (__bf16*)(ws + off); off += (size_t)N * DIN * 2;
  __bf16* hlo = (__bf16*)(ws + off);

  const int wslots = NREL * 8 * 4 * 32;
  k_split_weight<<<(wslots + 255) / 256, 256, 0, stream>>>(W, whi, wlo);
  const int tot_h = N * DIN;
  k_split_h<<<(tot_h + 255) / 256, 256, 0, stream>>>(h, hhi, hlo, tot_h);
  k_zero<<<(out_size + 255) / 256, 256, 0, stream>>>(out, out_size, deg, N);

  const int blocks = (E + CHUNK - 1) / CHUNK;
  k_rgcn<<<blocks, THREADS, 0, stream>>>(hhi, hlo, whi, wlo, src, dst, rel,
                                         out, deg, E);
  k_final<<<(out_size + 255) / 256, 256, 0, stream>>>(out, deg, bias, out_size);
}
